// AttnDecoderRNN_90348932039080
// MI455X (gfx1250) — compile-verified, matrix operand order checked by reading, not run on hardware
//
#include <hip/hip_runtime.h>
#include <hip/hip_bf16.h>

#define H    512
#define Bsz  256
#define S    512
#define Pdim 3
#define H3   1536
#define EPS  1e-5f

typedef __attribute__((ext_vector_type(16))) __bf16 v16bf;
typedef __attribute__((ext_vector_type(8)))  float  v8f;

static __device__ __forceinline__ v8f wmma_bf16(v16bf a, v16bf b, v8f c) {
  // D(16x16,f32) = A(16x32,bf16) * B(32x16,bf16) + C
  return __builtin_amdgcn_wmma_f32_16x16x32_bf16(false, a, false, b, (short)0, c,
                                                 false, false);
}

static __device__ __forceinline__ unsigned pk2(float lo, float hi) {
  unsigned short a = __builtin_bit_cast(unsigned short, (__bf16)lo);
  unsigned short b = __builtin_bit_cast(unsigned short, (__bf16)hi);
  return (unsigned)a | ((unsigned)b << 16);
}

// Fragment loader from f32 row-major source where each row holds K values.
// Layout (16-bit A 16x32): lanes 0-15 hold row (lane&15), K = 0..7,16..23;
// lanes 16-31 hold same rows with K = 8..15,24..31. B uses the mirrored layout.
static __device__ __forceinline__ v16bf frag_f32(const float* __restrict__ base,
                                                 int ld, int r0, int kc, int lane) {
  const int koff = (lane & 16) ? 8 : 0;
  const float* p = base + (size_t)(r0 + (lane & 15)) * ld + kc + koff;
  v16bf f;
#pragma unroll
  for (int i = 0; i < 8; ++i) { f[i] = (__bf16)p[i]; f[i + 8] = (__bf16)p[i + 16]; }
  return f;
}

static __device__ __forceinline__ v16bf frag_bf16(const __bf16* __restrict__ base,
                                                  int ld, int r0, int kc, int lane) {
  const int koff = (lane & 16) ? 8 : 0;
  const unsigned* q = (const unsigned*)(base + (size_t)(r0 + (lane & 15)) * ld + kc + koff);
  v16bf f; unsigned* fu = (unsigned*)&f;
#pragma unroll
  for (int i = 0; i < 4; ++i) { fu[i] = q[i]; fu[i + 4] = q[i + 8]; }
  return f;
}

// --- K0: pre-convert We (hot weight of the dominant GEMM) to bf16 ------------
__global__ void k_convert_we(const float* __restrict__ We, __bf16* __restrict__ weBf) {
  int i = (blockIdx.x * 256 + threadIdx.x) * 4;
#pragma unroll
  for (int j = 0; j < 4; ++j) weBf[i + j] = (__bf16)We[i + j];
}

// --- K1a: GRU gate GEMMs: gi = x@W_ih^T + b_ih, gh = h@W_hh^T + b_hh ---------
__global__ void __launch_bounds__(32) k_gru_gates(
    const float* __restrict__ palette, const float* __restrict__ lastCtx,
    const float* __restrict__ lastHid, const float* __restrict__ W_ih,
    const float* __restrict__ W_hh, const float* __restrict__ b_ih,
    const float* __restrict__ b_hh, float* __restrict__ gi, float* __restrict__ gh) {
  const int lane = threadIdx.x & 31;
  const int m0 = (blockIdx.x / 96) * 16;   // batch tile
  const int n0 = (blockIdx.x % 96) * 16;   // gate-feature tile (3H)
  v8f ai = {}; v8f ah = {};
  for (int kc = 0; kc < H; kc += 32) {
    v16bf a1 = frag_f32(lastCtx, H, m0, kc, lane);
    v16bf b1 = frag_f32(W_ih + Pdim, H + Pdim, n0, kc, lane);  // context columns
    ai = wmma_bf16(a1, b1, ai);
    v16bf a2 = frag_f32(lastHid, H, m0, kc, lane);
    v16bf b2 = frag_f32(W_hh, H, n0, kc, lane);
    ah = wmma_bf16(a2, b2, ah);
  }
  const int n = n0 + (lane & 15);
  const int mAdd = (lane & 16) ? 8 : 0;
  const float bi = b_ih[n], bhh = b_hh[n];
  const float w0 = W_ih[(size_t)n * (H + Pdim) + 0];
  const float w1 = W_ih[(size_t)n * (H + Pdim) + 1];
  const float w2 = W_ih[(size_t)n * (H + Pdim) + 2];
#pragma unroll
  for (int v = 0; v < 8; ++v) {
    int m = m0 + v + mAdd;
    float pv = palette[m * 3 + 0] * w0 + palette[m * 3 + 1] * w1 +
               palette[m * 3 + 2] * w2;                      // K=515 remainder
    gi[(size_t)m * H3 + n] = ai[v] + bi + pv;
    gh[(size_t)m * H3 + n] = ah[v] + bhh;
  }
}

// --- K1b: GRU elementwise combine -> h_new -----------------------------------
__global__ void k_gru_combine(const float* __restrict__ gi, const float* __restrict__ gh,
                              const float* __restrict__ lastHid, float* __restrict__ hNew) {
  int idx = blockIdx.x * 256 + threadIdx.x;  // over B*H
  int b = idx >> 9, h = idx & 511;
  const float* gib = gi + (size_t)b * H3;
  const float* ghb = gh + (size_t)b * H3;
  float r = 1.f / (1.f + __expf(-(gib[h] + ghb[h])));
  float z = 1.f / (1.f + __expf(-(gib[H + h] + ghb[H + h])));
  float n = tanhf(gib[2 * H + h] + r * ghb[2 * H + h]);
  hNew[idx] = (1.f - z) * n + z * lastHid[idx];
}

// --- K1c: hid_proj = h_new @ Wh^T + bh ---------------------------------------
__global__ void __launch_bounds__(32) k_hid_proj(
    const float* __restrict__ hNew, const float* __restrict__ Wh,
    const float* __restrict__ bh, float* __restrict__ hidProj) {
  const int lane = threadIdx.x & 31;
  const int m0 = (blockIdx.x / 32) * 16;
  const int n0 = (blockIdx.x % 32) * 16;
  v8f c = {};
  for (int kc = 0; kc < H; kc += 32) {
    v16bf a = frag_f32(hNew, H, m0, kc, lane);
    v16bf b = frag_f32(Wh, H, n0, kc, lane);
    c = wmma_bf16(a, b, c);
  }
  const int n = n0 + (lane & 15);
  const int mAdd = (lane & 16) ? 8 : 0;
  const float bias = bh[n];
#pragma unroll
  for (int v = 0; v < 8; ++v)
    hidProj[(size_t)(m0 + v + mAdd) * H + n] = c[v] + bias;
}

// --- K2: fused energy kernel (dominant). One block = one batch b, 64 s-rows. -
// energy[s,b] = sum_n sigmoid(enc[s,b,:]@We[n,:] + be[n] + hid_proj[b,n]) * Wv[n] + bv
// M dimension runs along s for a single b so that:
//   * early-out uses the true per-batch mask (s0 >= len[b]): ~50% skipped
//   * hid_proj/be/Wv are loop-invariant per lane
//   * each B-fragment (We tile) is reused across 4 M-tiles -> 4x less L2 traffic
__global__ void __launch_bounds__(512) k_energy(
    const float* __restrict__ enc, const __bf16* __restrict__ weBf,
    const float* __restrict__ hidProj, const float* __restrict__ be,
    const float* __restrict__ Wv, const float* __restrict__ bv,
    const int* __restrict__ lens, float* __restrict__ energy) {
  __shared__ unsigned tile[64 * 520 / 2];  // 64 rows x 512 bf16, +8 bf16 pad/row
  __shared__ float eng[64];
  const int tid = threadIdx.x;
  const int b = blockIdx.x >> 3;          // batch
  const int s0 = (blockIdx.x & 7) * 64;   // sequence chunk
  const int len = lens[b];
  if (s0 >= len) return;  // masked chunk: softmax weight is 0 anyway

  {  // stage 64 encoder rows enc[s0+r, b, :] into LDS as bf16
    int row = tid >> 3, cb = (tid & 7) * 64;
    const float* src = enc + ((size_t)(s0 + row) * Bsz + b) * H + cb;
    unsigned* dst = tile + row * 260 + (cb >> 1);
#pragma unroll
    for (int i = 0; i < 32; ++i) dst[i] = pk2(src[2 * i], src[2 * i + 1]);
  }
  if (tid < 64) eng[tid] = 0.f;
  __syncthreads();

  const int lane = tid & 31;
  const int wave = tid >> 5;  // 16 waves; wave owns N-tiles {2w, 2w+1}, all 4 M-tiles
  const int row = lane & 15;
  const int koff = (lane & 16) ? 8 : 0;
  v8f acc[4][2] = {};
  for (int kc = 0; kc < H; kc += 32) {
    v16bf bfr[2];
#pragma unroll
    for (int t = 0; t < 2; ++t)
      bfr[t] = frag_bf16(weBf, H, (wave * 2 + t) * 16, kc, lane);
#pragma unroll
    for (int mt = 0; mt < 4; ++mt) {
      const unsigned* p = tile + ((mt * 16 + row) * 520 + kc + koff) / 2;
      v16bf a; unsigned* au = (unsigned*)&a;
#pragma unroll
      for (int i = 0; i < 4; ++i) { au[i] = p[i]; au[i + 4] = p[i + 8]; }
#pragma unroll
      for (int t = 0; t < 2; ++t) acc[mt][t] = wmma_bf16(a, bfr[t], acc[mt][t]);
    }
  }
  const int mAdd = (lane & 16) ? 8 : 0;
  float rowAcc[4][8] = {};
#pragma unroll
  for (int t = 0; t < 2; ++t) {
    int n = (wave * 2 + t) * 16 + (lane & 15);
    float hb = be[n] + hidProj[(size_t)b * H + n];  // invariant over s
    float wv = Wv[n];
#pragma unroll
    for (int mt = 0; mt < 4; ++mt)
#pragma unroll
      for (int v = 0; v < 8; ++v) {
        float e = acc[mt][t][v] + hb;
        rowAcc[mt][v] += wv / (1.f + __expf(-e));
      }
  }
#pragma unroll
  for (int mt = 0; mt < 4; ++mt)
#pragma unroll
    for (int v = 0; v < 8; ++v) {  // reduce 16 lanes of each half (N direction)
      float x = rowAcc[mt][v];
      x += __shfl_xor(x, 1, 32);
      x += __shfl_xor(x, 2, 32);
      x += __shfl_xor(x, 4, 32);
      x += __shfl_xor(x, 8, 32);
      if ((lane & 15) == 0) atomicAdd(&eng[mt * 16 + v + mAdd], x);  // ds_add_f32
    }
  __syncthreads();
  if (tid < 64) energy[(size_t)(s0 + tid) * Bsz + b] = eng[tid] + bv[0];
}

// --- K3: masked softmax over S per column b; also emits attn_weights [B,1,S] -
__global__ void __launch_bounds__(128) k_softmax(
    const float* __restrict__ energy, const int* __restrict__ lens,
    float* __restrict__ aw, float* __restrict__ attnOut) {
  __shared__ float red[128];
  const int b = blockIdx.x, tid = threadIdx.x;
  const int len = lens[b];
  float ev[4], mx = -__builtin_inff();
#pragma unroll
  for (int i = 0; i < 4; ++i) {
    int s = tid + i * 128;
    float e = (s < len) ? energy[(size_t)s * Bsz + b] : -__builtin_inff();
    ev[i] = e; mx = fmaxf(mx, e);
  }
  red[tid] = mx; __syncthreads();
  for (int st = 64; st > 0; st >>= 1) {
    if (tid < st) red[tid] = fmaxf(red[tid], red[tid + st]);
    __syncthreads();
  }
  mx = red[0]; __syncthreads();
  float sum = 0.f;
#pragma unroll
  for (int i = 0; i < 4; ++i) {
    int s = tid + i * 128;
    if (s < len) sum += __expf(ev[i] - mx);
  }
  red[tid] = sum; __syncthreads();
  for (int st = 64; st > 0; st >>= 1) {
    if (tid < st) red[tid] += red[tid + st];
    __syncthreads();
  }
  float inv = 1.f / red[0];
#pragma unroll
  for (int i = 0; i < 4; ++i) {
    int s = tid + i * 128;
    float w = (s < len) ? __expf(ev[i] - mx) * inv : 0.f;
    aw[(size_t)s * Bsz + b] = w;
    attnOut[(size_t)b * S + s] = w;
  }
}

// --- K4: context[b,:] = sum_s aw[s,b] * enc[s,b,:], s bounded by len ---------
__global__ void __launch_bounds__(128) k_context(
    const float* __restrict__ enc, const float* __restrict__ aw,
    const int* __restrict__ lens, float* __restrict__ ctx) {
  __shared__ float sw[128];
  const int b = blockIdx.x, tid = threadIdx.x;
  const int len = lens[b];
  float4 acc = {0.f, 0.f, 0.f, 0.f};
  for (int s0 = 0; s0 < len; s0 += 128) {
    int cnt = min(128, len - s0);
    if (tid < cnt) sw[tid] = aw[(size_t)(s0 + tid) * Bsz + b];
    __syncthreads();
    for (int i = 0; i < cnt; ++i) {
      float w = sw[i];
      const float4 v = *(const float4*)(enc + ((size_t)((s0 + i) * Bsz + b)) * H + tid * 4);
      acc.x += w * v.x; acc.y += w * v.y; acc.z += w * v.z; acc.w += w * v.w;
    }
    __syncthreads();
  }
  *(float4*)(ctx + (size_t)b * H + tid * 4) = acc;
}

// --- K5a: y = relu([h_new | context] @ W1^T + b1) ----------------------------
__global__ void __launch_bounds__(32) k_mlp1(
    const float* __restrict__ hNew, const float* __restrict__ ctx,
    const float* __restrict__ W1, const float* __restrict__ b1,
    float* __restrict__ y) {
  const int lane = threadIdx.x & 31;
  const int m0 = (blockIdx.x / 32) * 16;
  const int n0 = (blockIdx.x % 32) * 16;
  v8f c = {};
  for (int kc = 0; kc < 2 * H; kc += 32) {
    const float* A = (kc < H) ? hNew : ctx;
    v16bf a = frag_f32(A, H, m0, kc & (H - 1), lane);
    v16bf b = frag_f32(W1, 2 * H, n0, kc, lane);
    c = wmma_bf16(a, b, c);
  }
  const int n = n0 + (lane & 15);
  const int mAdd = (lane & 16) ? 8 : 0;
  const float bias = b1[n];
#pragma unroll
  for (int v = 0; v < 8; ++v)
    y[(size_t)(m0 + v + mAdd) * H + n] = fmaxf(c[v] + bias, 0.f);
}

// --- K5b: batchnorm (train stats, biased var) over batch, per feature --------
__global__ void __launch_bounds__(256) k_bnorm(
    const float* __restrict__ y, const float* __restrict__ gamma,
    const float* __restrict__ beta, float* __restrict__ yn) {
  __shared__ float s1[256], s2[256], scl[64], sft[64];
  const int tid = threadIdx.x;
  const int fl = tid & 63, sl = tid >> 6;
  const int f = blockIdx.x * 64 + fl;
  float a = 0.f, bq = 0.f;
  for (int bb = sl; bb < Bsz; bb += 4) {
    float v = y[(size_t)bb * H + f];
    a += v; bq += v * v;
  }
  s1[tid] = a; s2[tid] = bq; __syncthreads();
  if (tid < 64) {
    float sa = s1[tid] + s1[tid + 64] + s1[tid + 128] + s1[tid + 192];
    float sb = s2[tid] + s2[tid + 64] + s2[tid + 128] + s2[tid + 192];
    float mu = sa * (1.f / Bsz);
    float var = sb * (1.f / Bsz) - mu * mu;
    float sc = rsqrtf(var + EPS) * gamma[blockIdx.x * 64 + tid];
    scl[tid] = sc;
    sft[tid] = beta[blockIdx.x * 64 + tid] - mu * sc;
  }
  __syncthreads();
  float sc = scl[fl], sh = sft[fl];
  for (int bb = sl; bb < Bsz; bb += 4)
    yn[(size_t)bb * H + f] = y[(size_t)bb * H + f] * sc + sh;
}

// --- K5c: out = yn @ W2^T + b2 (H=512 -> P=3) --------------------------------
__global__ void __launch_bounds__(128) k_out(
    const float* __restrict__ yn, const float* __restrict__ W2,
    const float* __restrict__ b2, float* __restrict__ out) {
  __shared__ float red[3][128];
  const int b = blockIdx.x, tid = threadIdx.x;
  float a0 = 0.f, a1 = 0.f, a2 = 0.f;
  for (int h = tid; h < H; h += 128) {
    float v = yn[(size_t)b * H + h];
    a0 += v * W2[h]; a1 += v * W2[H + h]; a2 += v * W2[2 * H + h];
  }
  red[0][tid] = a0; red[1][tid] = a1; red[2][tid] = a2; __syncthreads();
  for (int st = 64; st > 0; st >>= 1) {
    if (tid < st) {
      red[0][tid] += red[0][tid + st];
      red[1][tid] += red[1][tid + st];
      red[2][tid] += red[2][tid + st];
    }
    __syncthreads();
  }
  if (tid < 3) out[b * 3 + tid] = red[tid][0] + b2[tid];
}

extern "C" void kernel_launch(void* const* d_in, const int* in_sizes, int n_in,
                              void* d_out, int out_size, void* d_ws, size_t ws_size,
                              hipStream_t stream) {
  const float* palette = (const float*)d_in[0];
  const float* lastCtx = (const float*)d_in[1];
  const float* lastHid = (const float*)d_in[2];
  const float* enc     = (const float*)d_in[3];
  const int*   lens    = (const int*)d_in[4];
  const float* We   = (const float*)d_in[5];
  const float* be   = (const float*)d_in[6];
  const float* Wh   = (const float*)d_in[7];
  const float* bh   = (const float*)d_in[8];
  const float* Wv   = (const float*)d_in[9];
  const float* bv   = (const float*)d_in[10];
  const float* W_ih = (const float*)d_in[11];
  const float* W_hh = (const float*)d_in[12];
  const float* b_ih = (const float*)d_in[13];
  const float* b_hh = (const float*)d_in[14];
  const float* W1   = (const float*)d_in[15];
  const float* b1   = (const float*)d_in[16];
  const float* gam  = (const float*)d_in[17];
  const float* bet  = (const float*)d_in[18];
  const float* W2   = (const float*)d_in[19];
  const float* b2   = (const float*)d_in[20];

  // d_out: out[B,P] | context[B,H] | h_new[B,H] | attn[B,1,S]
  float* out  = (float*)d_out;
  float* ctx  = out + Bsz * Pdim;
  float* hNew = ctx + Bsz * H;
  float* attn = hNew + Bsz * H;

  // workspace layout (floats)
  float* ws      = (float*)d_ws;
  float* gi      = ws;                   // B*3H
  float* gh      = gi + Bsz * H3;        // B*3H
  float* hidProj = gh + Bsz * H3;        // B*H
  float* energy  = hidProj + Bsz * H;    // S*B
  float* aw      = energy + S * Bsz;     // S*B
  float* y       = aw + S * Bsz;         // B*H
  float* yn      = y + Bsz * H;          // B*H
  __bf16* weBf   = (__bf16*)(yn + Bsz * H);  // H*H bf16

  k_convert_we <<<256, 256, 0, stream>>>(We, weBf);
  k_gru_gates  <<<16 * 96, 32, 0, stream>>>(palette, lastCtx, lastHid, W_ih, W_hh,
                                            b_ih, b_hh, gi, gh);
  k_gru_combine<<<(Bsz * H) / 256, 256, 0, stream>>>(gi, gh, lastHid, hNew);
  k_hid_proj   <<<16 * 32, 32, 0, stream>>>(hNew, Wh, bh, hidProj);
  k_energy     <<<Bsz * (S / 64), 512, 0, stream>>>(enc, weBf, hidProj, be, Wv, bv,
                                                    lens, energy);
  k_softmax    <<<Bsz, 128, 0, stream>>>(energy, lens, aw, attn);
  k_context    <<<Bsz, 128, 0, stream>>>(enc, aw, lens, ctx);
  k_mlp1       <<<16 * 32, 32, 0, stream>>>(hNew, ctx, W1, b1, y);
  k_bnorm      <<<H / 64, 256, 0, stream>>>(y, gam, bet, yn);
  k_out        <<<Bsz, 128, 0, stream>>>(yn, W2, b2, out);
}
